// MultiHeadAttention_80925773791779
// MI455X (gfx1250) — matrix operand order repaired and checked by reading, not compiled, not run on hardware
//
#include <hip/hip_runtime.h>

typedef __attribute__((ext_vector_type(16))) _Float16 v16h;
typedef __attribute__((ext_vector_type(8)))  _Float16 v8h;
typedef __attribute__((ext_vector_type(4)))  _Float16 v4h;
typedef __attribute__((ext_vector_type(8)))  float    v8f;
typedef __attribute__((ext_vector_type(4)))  float    v4f;

namespace {
constexpr int Bc = 2, Tc = 2048, Dc = 1024, Hc = 16, HDc = 64;
constexpr int XN = Bc * Tc * Dc;        // 4,194,304 x elements
constexpr int WN = Hc * HDc * Dc;       // 1,048,576 per-weight elements
constexpr int PN = Dc * Dc;             // 1,048,576 Wp elements
}

__device__ __forceinline__ v8f wmma_f16(v16h a, v16h b, v8f c) {
  // D = A(16x32 f16) * B(32x16 f16) + C(16x16 f32)
  return __builtin_amdgcn_wmma_f32_16x16x32_f16(false, a, false, b, (short)0, c,
                                                false, false);
}

__device__ __forceinline__ v16h cat8(v8h lo, v8h hi) {
  return __builtin_shufflevector(lo, hi, 0, 1, 2, 3, 4, 5, 6, 7, 8, 9, 10, 11,
                                 12, 13, 14, 15);
}

// ---------------------------------------------------------------- converts
// 4-wide: b128 load, packed cvt, b64 store. n is always a multiple of 4.
__global__ void f32_to_f16_kernel(const float* __restrict__ src,
                                  _Float16* __restrict__ dst, int n4) {
  int i = blockIdx.x * blockDim.x + threadIdx.x;
  if (i < n4) {
    v4f s = *(const v4f*)(src + 4 * (size_t)i);
    v4h d;
    d[0] = (_Float16)s[0]; d[1] = (_Float16)s[1];
    d[2] = (_Float16)s[2]; d[3] = (_Float16)s[3];
    *(v4h*)(dst + 4 * (size_t)i) = d;
  }
}

// ---------------------------------------------------------------- QKV GEMM
// One wave computes a 32(t) x 64(e) tile of q, k, or v for one (b,h):
// 2 A-frags x 4 B-frags -> 8 WMMAs per 32-wide k-step (~21 FLOP/B of L2).
// q,k : [B,H,T,HD] row-major (f16).  vT : [B,H,HD,T] (f16, transposed store).
__global__ __launch_bounds__(128) void qkv_gemm_kernel(
    const _Float16* __restrict__ xh, const _Float16* __restrict__ Wqh,
    const _Float16* __restrict__ Wkh, const _Float16* __restrict__ Wvh,
    _Float16* __restrict__ q, _Float16* __restrict__ k,
    _Float16* __restrict__ vT) {
  const int lane = threadIdx.x & 31;
  const int wid = blockIdx.x * (blockDim.x >> 5) + (threadIdx.x >> 5);
  constexpr int TB = Tc / 32;                    // 64 t-blocks
  constexpr int PER_MAT = Bc * Hc * TB;          // 2048

  const int mat = wid / PER_MAT;                 // 0=q, 1=k, 2=v
  const int rem = wid % PER_MAT;
  const int bh = rem / TB;
  const int t0 = (rem % TB) * 32;
  const int b = bh >> 4, h = bh & 15;

  const _Float16* W = (mat == 0) ? Wqh : (mat == 1) ? Wkh : Wvh;

  const int half = lane >> 4, l16 = lane & 15;
  const int koffA = half * 16;   // A-frag K sub-offset per half-wave
  const int krowB = half * 16;   // B-frag K base per half-wave

  const _Float16* arow0 = xh + (size_t)(b * Tc + t0 + l16) * Dc;
  const _Float16* arow1 = arow0 + (size_t)16 * Dc;
  const _Float16* brow0 = W + (size_t)(h * HDc + 0 + l16) * Dc;
  const _Float16* brow1 = W + (size_t)(h * HDc + 16 + l16) * Dc;
  const _Float16* brow2 = W + (size_t)(h * HDc + 32 + l16) * Dc;
  const _Float16* brow3 = W + (size_t)(h * HDc + 48 + l16) * Dc;

  v8f acc[2][4];
#pragma unroll
  for (int mt = 0; mt < 2; ++mt)
#pragma unroll
    for (int et = 0; et < 4; ++et) acc[mt][et] = (v8f){};

#pragma unroll 2
  for (int k0 = 0; k0 < Dc; k0 += 32) {
    if (k0 + 256 < Dc) {  // prefetch next K panel of A rows
      __builtin_prefetch(arow0 + k0 + 256, 0, 3);
      __builtin_prefetch(arow1 + k0 + 256, 0, 3);
    }
    v16h af[2];
    af[0] = cat8(*(const v8h*)(arow0 + k0 + koffA),
                 *(const v8h*)(arow0 + k0 + 8 + koffA));
    af[1] = cat8(*(const v8h*)(arow1 + k0 + koffA),
                 *(const v8h*)(arow1 + k0 + 8 + koffA));
    v16h bf[4];
    bf[0] = *(const v16h*)(brow0 + k0 + krowB);
    bf[1] = *(const v16h*)(brow1 + k0 + krowB);
    bf[2] = *(const v16h*)(brow2 + k0 + krowB);
    bf[3] = *(const v16h*)(brow3 + k0 + krowB);
#pragma unroll
    for (int mt = 0; mt < 2; ++mt)
#pragma unroll
      for (int et = 0; et < 4; ++et)
        acc[mt][et] = wmma_f16(af[mt], bf[et], acc[mt][et]);
  }

  if (mat == 2) {
    // transposed store: vT[bh*HD + e][t] ; 8 contiguous t per lane per tile
#pragma unroll
    for (int mt = 0; mt < 2; ++mt)
#pragma unroll
      for (int et = 0; et < 4; ++et) {
        v8h o;
#pragma unroll
        for (int r = 0; r < 8; ++r) o[r] = (_Float16)acc[mt][et][r];
        *(v8h*)(vT + (size_t)(bh * HDc + et * 16 + l16) * Tc + t0 + mt * 16 +
                half * 8) = o;
      }
  } else {
    _Float16* dst = (mat == 0) ? q : k;
#pragma unroll
    for (int mt = 0; mt < 2; ++mt)
#pragma unroll
      for (int et = 0; et < 4; ++et)
#pragma unroll
        for (int r = 0; r < 8; ++r)
          dst[(size_t)(bh * Tc + t0 + mt * 16 + half * 8 + r) * HDc + et * 16 +
              l16] = (_Float16)acc[mt][et][r];
  }
}

// ---------------------------------------------------------------- attention
// One wave per (b, h, 16-row block). Streaming (flash) softmax over s-chunks
// of 32 with online max/sum; causal mask; output to att[B,T,D] (f16).
__global__ __launch_bounds__(32) void attn_kernel(
    const _Float16* __restrict__ q, const _Float16* __restrict__ k,
    const _Float16* __restrict__ vT, _Float16* __restrict__ att) {
  __shared__ _Float16 P[16 * 40];  // 16x32 P-tile, padded row stride 40

  const int lane = threadIdx.x;
  const int bh = blockIdx.x >> 7;          // /128 row-blocks
  const int t0 = (blockIdx.x & 127) * 16;
  const int b = bh >> 4, h = bh & 15;

  const int half = lane >> 4, l16 = lane & 15;
  const int koffA = half * 16;
  const int krowB = half * 16;

  // Q A-fragments (K = e in [0,64)), reused across all s-chunks
  const _Float16* qrow = q + (size_t)(bh * Tc + t0 + l16) * HDc;
  const v16h a0 = cat8(*(const v8h*)(qrow + koffA),
                       *(const v8h*)(qrow + 8 + koffA));
  const v16h a1 = cat8(*(const v8h*)(qrow + 32 + koffA),
                       *(const v8h*)(qrow + 40 + koffA));

  float m_i[8], l_i[8];
  v8f o0 = {}, o1 = {}, o2 = {}, o3 = {};
#pragma unroll
  for (int r = 0; r < 8; ++r) { m_i[r] = -1e30f; l_i[r] = 0.0f; }

  const float inv_sqrt_hd = 0.125f;  // 1/sqrt(64)

  for (int s0 = 0; s0 <= t0 + 15; s0 += 32) {
    if (s0 + 32 <= t0 + 15) {  // prefetch next chunk's K rows
      __builtin_prefetch(k + (size_t)(bh * Tc + s0 + 32 + l16) * HDc, 0, 3);
      __builtin_prefetch(k + (size_t)(bh * Tc + s0 + 48 + l16) * HDc, 0, 3);
    }
    // ---- scores: two 16x16 tiles (s-cols s0..s0+15, s0+16..s0+31)
    v8f sa[2];
#pragma unroll
    for (int j = 0; j < 2; ++j) {
      const _Float16* krow =
          k + (size_t)(bh * Tc + s0 + j * 16 + l16) * HDc;
      v8f sc = {};
      sc = wmma_f16(a0, *(const v16h*)(krow + krowB), sc);
      sc = wmma_f16(a1, *(const v16h*)(krow + 32 + krowB), sc);
      sa[j] = sc;
    }

    // ---- online softmax update (per accumulator row r)
#pragma unroll
    for (int r = 0; r < 8; ++r) {
      const int row = t0 + half * 8 + r;
      float v0 = sa[0][r] * inv_sqrt_hd;
      float v1 = sa[1][r] * inv_sqrt_hd;
      if (s0 + l16 > row) v0 = -1e30f;        // causal mask
      if (s0 + 16 + l16 > row) v1 = -1e30f;
      float cm = fmaxf(v0, v1);
      cm = fmaxf(cm, __shfl_xor(cm, 1));
      cm = fmaxf(cm, __shfl_xor(cm, 2));
      cm = fmaxf(cm, __shfl_xor(cm, 4));
      cm = fmaxf(cm, __shfl_xor(cm, 8));      // row max over 16 lanes
      const float mnew = fmaxf(m_i[r], cm);
      const float corr = __expf(m_i[r] - mnew);
      const float p0 = __expf(v0 - mnew);
      const float p1 = __expf(v1 - mnew);
      float ps = p0 + p1;
      ps += __shfl_xor(ps, 1);
      ps += __shfl_xor(ps, 2);
      ps += __shfl_xor(ps, 4);
      ps += __shfl_xor(ps, 8);                // row sum over 16 lanes
      l_i[r] = l_i[r] * corr + ps;
      m_i[r] = mnew;
      o0[r] *= corr; o1[r] *= corr; o2[r] *= corr; o3[r] *= corr;
      P[(half * 8 + r) * 40 + l16] = (_Float16)p0;
      P[(half * 8 + r) * 40 + 16 + l16] = (_Float16)p1;
    }
    __syncthreads();  // order LDS P writes vs cross-lane A-frag reads

    // ---- reshape P (C-layout) -> A-fragment via LDS
    const _Float16* prow = &P[l16 * 40];
    const v16h pa = cat8(*(const v8h*)(prow + koffA),
                         *(const v8h*)(prow + 8 + koffA));

    // ---- out += P @ V  (B-frags contiguous from vT[e][s])
    {
      const size_t vbase = (size_t)bh * HDc * Tc + s0 + krowB;
      o0 = wmma_f16(pa, *(const v16h*)(vT + vbase + (size_t)(0 + l16) * Tc), o0);
      o1 = wmma_f16(pa, *(const v16h*)(vT + vbase + (size_t)(16 + l16) * Tc), o1);
      o2 = wmma_f16(pa, *(const v16h*)(vT + vbase + (size_t)(32 + l16) * Tc), o2);
      o3 = wmma_f16(pa, *(const v16h*)(vT + vbase + (size_t)(48 + l16) * Tc), o3);
    }
    __syncthreads();  // protect P before next iteration overwrites it
  }

  // ---- normalize and store to att[B,T,D] (heads concatenated)
#pragma unroll
  for (int r = 0; r < 8; ++r) {
    const float inv = 1.0f / l_i[r];
    const size_t rowbase = (size_t)(b * Tc + t0 + half * 8 + r) * Dc + h * HDc;
    att[rowbase + 0 + l16] = (_Float16)(o0[r] * inv);
    att[rowbase + 16 + l16] = (_Float16)(o1[r] * inv);
    att[rowbase + 32 + l16] = (_Float16)(o2[r] * inv);
    att[rowbase + 48 + l16] = (_Float16)(o3[r] * inv);
  }
}

// ---------------------------------------------------------------- out proj
// y[m,n] = sum_d att[m,d] * Wp[n,d] + bp[n], f32 output.
// One wave computes a 32x64 tile: 8 WMMAs per 32-wide k-step.
__global__ __launch_bounds__(128) void proj_kernel(
    const _Float16* __restrict__ att, const _Float16* __restrict__ Wph,
    const float* __restrict__ bp, float* __restrict__ y) {
  const int lane = threadIdx.x & 31;
  const int wid = blockIdx.x * (blockDim.x >> 5) + (threadIdx.x >> 5);
  constexpr int TILES_N = Dc / 64;  // 16 column groups
  const int m0 = (wid / TILES_N) * 32;
  const int n0 = (wid % TILES_N) * 64;

  const int half = lane >> 4, l16 = lane & 15;
  const int koffA = half * 16;
  const int krowB = half * 16;

  const _Float16* arow0 = att + (size_t)(m0 + l16) * Dc;
  const _Float16* arow1 = arow0 + (size_t)16 * Dc;
  const _Float16* brow0 = Wph + (size_t)(n0 + 0 + l16) * Dc;
  const _Float16* brow1 = Wph + (size_t)(n0 + 16 + l16) * Dc;
  const _Float16* brow2 = Wph + (size_t)(n0 + 32 + l16) * Dc;
  const _Float16* brow3 = Wph + (size_t)(n0 + 48 + l16) * Dc;

  v8f acc[2][4];
#pragma unroll
  for (int mt = 0; mt < 2; ++mt)
#pragma unroll
    for (int et = 0; et < 4; ++et) acc[mt][et] = (v8f){};

#pragma unroll 2
  for (int k0 = 0; k0 < Dc; k0 += 32) {
    if (k0 + 256 < Dc) {
      __builtin_prefetch(arow0 + k0 + 256, 0, 3);
      __builtin_prefetch(arow1 + k0 + 256, 0, 3);
    }
    v16h af[2];
    af[0] = cat8(*(const v8h*)(arow0 + k0 + koffA),
                 *(const v8h*)(arow0 + k0 + 8 + koffA));
    af[1] = cat8(*(const v8h*)(arow1 + k0 + koffA),
                 *(const v8h*)(arow1 + k0 + 8 + koffA));
    v16h bf[4];
    bf[0] = *(const v16h*)(brow0 + k0 + krowB);
    bf[1] = *(const v16h*)(brow1 + k0 + krowB);
    bf[2] = *(const v16h*)(brow2 + k0 + krowB);
    bf[3] = *(const v16h*)(brow3 + k0 + krowB);
#pragma unroll
    for (int mt = 0; mt < 2; ++mt)
#pragma unroll
      for (int et = 0; et < 4; ++et)
        acc[mt][et] = wmma_f16(af[mt], bf[et], acc[mt][et]);
  }

#pragma unroll
  for (int et = 0; et < 4; ++et) {
    const float bias = bp[n0 + et * 16 + l16];
#pragma unroll
    for (int mt = 0; mt < 2; ++mt)
#pragma unroll
      for (int r = 0; r < 8; ++r)
        y[(size_t)(m0 + mt * 16 + half * 8 + r) * Dc + n0 + et * 16 + l16] =
            acc[mt][et][r] + bias;
  }
}

// ---------------------------------------------------------------- launcher
extern "C" void kernel_launch(void* const* d_in, const int* in_sizes, int n_in,
                              void* d_out, int out_size, void* d_ws,
                              size_t ws_size, hipStream_t stream) {
  const float* x = (const float*)d_in[0];
  const float* Wq = (const float*)d_in[1];
  const float* Wk = (const float*)d_in[2];
  const float* Wv = (const float*)d_in[3];
  const float* Wp = (const float*)d_in[4];
  const float* bp = (const float*)d_in[5];
  float* y = (float*)d_out;

  // workspace layout (f16, bytes): 48 MB total
  char* ws = (char*)d_ws;
  _Float16* xh = (_Float16*)(ws);                          // 8 MB
  _Float16* Wqh = (_Float16*)(ws + 8388608);               // 2 MB
  _Float16* Wkh = (_Float16*)(ws + 10485760);              // 2 MB
  _Float16* Wvh = (_Float16*)(ws + 12582912);              // 2 MB
  _Float16* Wph = (_Float16*)(ws + 14680064);              // 2 MB
  _Float16* q = (_Float16*)(ws + 16777216);                // 8 MB
  _Float16* k = (_Float16*)(ws + 25165824);                // 8 MB
  _Float16* vT = (_Float16*)(ws + 33554432);               // 8 MB
  _Float16* att = (_Float16*)(ws + 41943040);              // 8 MB

  // 1) convert inputs to f16 (4-wide vectorized)
  f32_to_f16_kernel<<<(XN / 4 + 255) / 256, 256, 0, stream>>>(x, xh, XN / 4);
  f32_to_f16_kernel<<<(WN / 4 + 255) / 256, 256, 0, stream>>>(Wq, Wqh, WN / 4);
  f32_to_f16_kernel<<<(WN / 4 + 255) / 256, 256, 0, stream>>>(Wk, Wkh, WN / 4);
  f32_to_f16_kernel<<<(WN / 4 + 255) / 256, 256, 0, stream>>>(Wv, Wvh, WN / 4);
  f32_to_f16_kernel<<<(PN / 4 + 255) / 256, 256, 0, stream>>>(Wp, Wph, PN / 4);

  // 2) QKV projections: 3 * 2048 (32x64)-tiles, 4 waves/block
  qkv_gemm_kernel<<<1536, 128, 0, stream>>>(xh, Wqh, Wkh, Wvh, q, k, vT);

  // 3) flash attention: B*H*(T/16) = 4096 single-wave blocks
  attn_kernel<<<4096, 32, 0, stream>>>(q, k, vT, att);

  // 4) output projection + bias: 2048 (32x64)-tiles, 4 waves/block
  proj_kernel<<<512, 128, 0, stream>>>(att, Wph, bp, y);

  (void)in_sizes; (void)n_in; (void)out_size; (void)ws_size;
}
